// KANLinear_20864951124629
// MI455X (gfx1250) — matrix operand order repaired and checked by reading, not compiled, not run on hardware
//
#include <hip/hip_runtime.h>
#include <hip/hip_bf16.h>

typedef __attribute__((ext_vector_type(16))) __bf16 bf16x16;
typedef __attribute__((ext_vector_type(8)))  float  v8f;
typedef __attribute__((ext_vector_type(4)))  unsigned int u32x4;
typedef __attribute__((ext_vector_type(8)))  int    i32x8;
typedef __attribute__((ext_vector_type(4)))  int    i32x4;

#define BM 128
#define BN 128
#define BK 32
#define KTOT 9216          // 1024 (silu) + 8192 (basis)
#define LPITCH 40          // padded LDS row (bf16 elems): 80B -> conflict-free frag reads

static __device__ __forceinline__ __bf16 tobf(float f) { return (__bf16)f; }

static __device__ __forceinline__ float silu(float v) {
    return v / (1.0f + __expf(-v));
}

// Uniform cubic B-spline basis, knots g[j] = -2.2 + 0.4*j (j=0..11), outputs B_0..B_7.
// Exact closed form of the reference Cox-de Boor recurrence on a uniform knot vector.
static __device__ __forceinline__ void basis8(float x, __bf16* out) {
    float u  = (x + 2.2f) * 2.5f;       // (x - g0)/h
    float fu = floorf(u);
    int   i  = (int)fu;                 // interval index, 0..10 when valid
    float t  = u - fu;
    bool  valid = (x >= -2.2f) && (x < 2.2f);
    float omt = 1.0f - t;
    float t2 = t * t, t3 = t2 * t;
    const float k6 = 1.0f / 6.0f;
    float c0 = omt * omt * omt * k6;                            // B_{i-3}
    float c1 = (3.0f * t3 - 6.0f * t2 + 4.0f) * k6;             // B_{i-2}
    float c2 = (-3.0f * t3 + 3.0f * t2 + 3.0f * t + 1.0f) * k6; // B_{i-1}
    float c3 = t3 * k6;                                         // B_i
#pragma unroll
    for (int j = 0; j < 8; ++j) {
        int d = i - j;
        float v = (d == 3) ? c0 : (d == 2) ? c1 : (d == 1) ? c2 : (d == 0) ? c3 : 0.0f;
        out[j] = tobf(valid ? v : 0.0f);
    }
}

// Issue a TDM 2D tile load: 128 rows x 64B from Wcat (row stride 18432B) into LDS,
// with 16B padding after every 64B row -> 80B LDS pitch (matches LPITCH).
static __device__ __forceinline__ void tdm_load_b_tile(const __bf16* gsrc, unsigned lds_off) {
    unsigned long long ga = (unsigned long long)(size_t)gsrc;
    u32x4 g0;
    g0[0] = 1u;                                              // count=1 (valid), no gather
    g0[1] = lds_off;                                         // lds_addr (bytes)
    g0[2] = (unsigned)(ga & 0xffffffffu);                    // global_addr[31:0]
    g0[3] = (unsigned)((ga >> 32) & 0x01ffffffu) | (2u << 30); // global_addr[56:32] | type=2
    i32x8 g1;
    g1[0] = (1 << 16)        // data_size = 2 bytes
          | (1 << 20)        // pad_enable
          | (3 << 22)        // pad_interval: code 3 = 16 DWORDs = 64B
          | (3 << 25);       // pad_amount:   code 3 = 4 DWORDs = 16B
    g1[1] = (int)(9216u << 16);   // tensor_dim0[15:0] (bits 63:48)
    g1[2] = (int)(1024u << 16);   // tensor_dim0 hi = 0, tensor_dim1[15:0] = 1024
    g1[3] = (int)(32u << 16);     // tensor_dim1 hi = 0, tile_dim0 = 32 elems (64B)
    g1[4] = 128;                  // tile_dim1 = 128 rows, tile_dim2 = 0 (2D)
    g1[5] = 9216;                 // tensor_dim0_stride[31:0] (elements)
    g1[6] = 0;
    g1[7] = 0;
    i32x4 z4 = {0, 0, 0, 0};
    i32x8 z8 = {0, 0, 0, 0, 0, 0, 0, 0};
    __builtin_amdgcn_tensor_load_to_lds(g0, g1, z4, z4, z8, 0);
}

// Pack Wcat bf16 [1024][9216]: cols [0,1024) = base_weight, cols [1024,9216) = spline_weight*scaler
__global__ __launch_bounds__(256) void kan_pack_w(const float* __restrict__ bw,
                                                  const float* __restrict__ sw,
                                                  const float* __restrict__ sc,
                                                  __bf16* __restrict__ wcat) {
    int idx = blockIdx.x * 256 + threadIdx.x;   // o*1024 + f
    int o = idx >> 10, f = idx & 1023;
    __bf16* wrow = wcat + (size_t)o * KTOT;
    wrow[f] = tobf(bw[idx]);
    float s = sc[idx];
    const float* swp = sw + (size_t)idx * 8;
    __bf16 tmp[8];
#pragma unroll
    for (int c = 0; c < 8; ++c) tmp[c] = tobf(swp[c] * s);
    *(uint4*)&wrow[1024 + f * 8] = *(const uint4*)tmp;
}

// Fused GEMM: out[b][o] = sum_k Acat[b][k] * Wcat[o][k], Acat generated on the fly.
__global__ __launch_bounds__(256) void kan_gemm(const float* __restrict__ x,
                                                const __bf16* __restrict__ wcat,
                                                float* __restrict__ out) {
    __shared__ __bf16 Alds[BM][LPITCH];
    __shared__ __bf16 Blds[BN][LPITCH];

    const int tid  = threadIdx.x;
    const int lane = tid & 31;
    const int wave = tid >> 5;
    const int wm   = wave >> 2;          // 2 waves in M (64 rows each)
    const int wn   = wave & 3;           // 4 waves in N (32 cols each)
    const int half = lane >> 4;
    const int l15  = lane & 15;

    const int row0 = blockIdx.y * BM;
    const int col0 = blockIdx.x * BN;

    v8f acc[4][2] = {};

    const int fr = tid >> 1;             // A fill: row, 2 threads/row
    const int fc = (tid & 1) * 16;       // 16-elem column half

    const unsigned blds_off = (unsigned)(size_t)&Blds[0][0];  // LDS byte offset (low 32 bits)
    const __bf16* wrow = wcat + (size_t)col0 * KTOT;          // first B row of this block

    union Frag { bf16x16 v; uint4 u[2]; };

    // ---------------- Phase 1: K in [0,1024) — A = silu(x) ----------------
    for (int kt = 0; kt < 1024; kt += BK) {
        if (wave == 0) tdm_load_b_tile(wrow + kt, blds_off);
        {
            const float4* px = (const float4*)(x + (size_t)(row0 + fr) * 1024 + kt + fc);
            __bf16 tmp[16];
#pragma unroll
            for (int q = 0; q < 4; ++q) {
                float4 v = px[q];
                tmp[q * 4 + 0] = tobf(silu(v.x));
                tmp[q * 4 + 1] = tobf(silu(v.y));
                tmp[q * 4 + 2] = tobf(silu(v.z));
                tmp[q * 4 + 3] = tobf(silu(v.w));
            }
            *(uint4*)&Alds[fr][fc]     = *(const uint4*)&tmp[0];
            *(uint4*)&Alds[fr][fc + 8] = *(const uint4*)&tmp[8];
        }
        if (wave == 0) __builtin_amdgcn_s_wait_tensorcnt(0);
        __syncthreads();

        Frag af[4], bfg[2];
#pragma unroll
        for (int mi = 0; mi < 4; ++mi) {
            int arow = wm * 64 + mi * 16 + l15;
            af[mi].u[0] = *(const uint4*)&Alds[arow][half * 8];
            af[mi].u[1] = *(const uint4*)&Alds[arow][16 + half * 8];
        }
#pragma unroll
        for (int ni = 0; ni < 2; ++ni) {
            int brow = wn * 32 + ni * 16 + l15;
            bfg[ni].u[0] = *(const uint4*)&Blds[brow][half * 8];
            bfg[ni].u[1] = *(const uint4*)&Blds[brow][16 + half * 8];
        }
#pragma unroll
        for (int mi = 0; mi < 4; ++mi)
#pragma unroll
            for (int ni = 0; ni < 2; ++ni)
                acc[mi][ni] = __builtin_amdgcn_wmma_f32_16x16x32_bf16(
                    false, af[mi].v, false, bfg[ni].v,
                    (short)0, acc[mi][ni], false, false);
        __syncthreads();
    }

    // ---------------- Phase 2: K in [1024,9216) — A = basis(x) ----------------
    for (int kt = 0; kt < 8192; kt += BK) {
        if (wave == 0) tdm_load_b_tile(wrow + 1024 + kt, blds_off);
        {
            const int f0 = kt >> 3;      // 4 features per BK=32 slab
#pragma unroll
            for (int e = tid; e < 512; e += 256) {
                int r  = e >> 2;
                int fi = e & 3;
                float xv = x[(size_t)(row0 + r) * 1024 + f0 + fi];
                __bf16 tmp[8];
                basis8(xv, tmp);
                *(uint4*)&Alds[r][fi * 8] = *(const uint4*)tmp;
            }
        }
        if (wave == 0) __builtin_amdgcn_s_wait_tensorcnt(0);
        __syncthreads();

        Frag af[4], bfg[2];
#pragma unroll
        for (int mi = 0; mi < 4; ++mi) {
            int arow = wm * 64 + mi * 16 + l15;
            af[mi].u[0] = *(const uint4*)&Alds[arow][half * 8];
            af[mi].u[1] = *(const uint4*)&Alds[arow][16 + half * 8];
        }
#pragma unroll
        for (int ni = 0; ni < 2; ++ni) {
            int brow = wn * 32 + ni * 16 + l15;
            bfg[ni].u[0] = *(const uint4*)&Blds[brow][half * 8];
            bfg[ni].u[1] = *(const uint4*)&Blds[brow][16 + half * 8];
        }
#pragma unroll
        for (int mi = 0; mi < 4; ++mi)
#pragma unroll
            for (int ni = 0; ni < 2; ++ni)
                acc[mi][ni] = __builtin_amdgcn_wmma_f32_16x16x32_bf16(
                    false, af[mi].v, false, bfg[ni].v,
                    (short)0, acc[mi][ni], false, false);
        __syncthreads();
    }

    // ---- store C: VGPR r -> M = r + 8*half, N = l15
#pragma unroll
    for (int mi = 0; mi < 4; ++mi) {
#pragma unroll
        for (int ni = 0; ni < 2; ++ni) {
            int orow = row0 + wm * 64 + mi * 16 + half * 8;
            int ocol = col0 + wn * 32 + ni * 16 + l15;
            float* po = out + (size_t)orow * 1024 + ocol;
#pragma unroll
            for (int r = 0; r < 8; ++r)
                po[(size_t)r * 1024] = acc[mi][ni][r];
        }
    }
}

extern "C" void kernel_launch(void* const* d_in, const int* in_sizes, int n_in,
                              void* d_out, int out_size, void* d_ws, size_t ws_size,
                              hipStream_t stream) {
    (void)in_sizes; (void)n_in; (void)out_size; (void)ws_size;
    const float* x  = (const float*)d_in[0];
    const float* bw = (const float*)d_in[1];
    const float* sw = (const float*)d_in[2];
    const float* sc = (const float*)d_in[3];
    // d_in[4] = grid: uniform, hardcoded (h=0.4, g0=-2.2)
    float* out = (float*)d_out;
    __bf16* wcat = (__bf16*)d_ws;   // 1024*9216 bf16 = 18 MB

    kan_pack_w<<<(1024 * 1024) / 256, 256, 0, stream>>>(bw, sw, sc, wcat);
    dim3 grid(1024 / BN, 8192 / BM);   // (8, 64)
    kan_gemm<<<grid, 256, 0, stream>>>(x, wcat, out);
}
